// SpatialCrossAttention_31009663877594
// MI455X (gfx1250) — matrix operand order repaired and checked by reading, not compiled, not run on hardware
//
#include <hip/hip_runtime.h>
#include <hip/hip_bf16.h>
#include <math.h>

typedef __attribute__((ext_vector_type(16))) _Float16 v16h;
typedef __attribute__((ext_vector_type(8)))  float    v8f;
typedef __attribute__((ext_vector_type(4)))  unsigned v4u;
typedef __attribute__((ext_vector_type(8)))  unsigned v8u;

namespace {

constexpr int kB = 4, kC = 256, kH = 32, kW = 32;
constexpr int kG = 4, kHEADS = 16, kDH = 16, kCG = 64;
constexpr int kDF = 4, kKOFF = 6;
constexpr int kH2 = 8, kW2 = 8, kNJ = 64;          // h2, w2, h2*w2
constexpr int kHW = 1024;
constexpr int kHPG = 4, kCPB = 64;
constexpr float kOSCALE = 4.0f;

__device__ __forceinline__ v8f wmma16x16x32(v16h a, v16h b, v8f c) {
  return __builtin_amdgcn_wmma_f32_16x16x32_f16(false, a, false, b, (short)0, c,
                                                false, false);
}

// ---- fragment loaders (wave32 WMMA; A/B use contiguous K order, not the ISA table order) -------------
// A: 16x32 f16. lane L: row = L&15; half i -> K = (L>>4)*16 + i
__device__ __forceinline__ v16h afrag(const float* A, int lda, int row0, int k0,
                                      int lane) {
  v16h a;
  const float* ar = A + (size_t)(row0 + (lane & 15)) * lda + k0 + ((lane >> 4) * 16);
#pragma unroll
  for (int i = 0; i < 16; ++i) a[i] = (_Float16)ar[i];
  return a;
}

__device__ __forceinline__ v16h afrag_h(const _Float16* A, int lda, int row0,
                                        int k0, int lane) {
  v16h a;
  const _Float16* ar = A + (size_t)(row0 + (lane & 15)) * lda + k0 + ((lane >> 4) * 16);
#pragma unroll
  for (int i = 0; i < 16; ++i) a[i] = ar[i];
  return a;
}

// transposed A source: A[m][kk] = At[kk*ld + row0 + m]; masked K + scale
__device__ __forceinline__ v16h afragT_ms(const float* At, int ld, int row0,
                                          int k0, int kmax, float scale, int lane) {
  v16h a;
  int row = row0 + (lane & 15);
  int koff = k0 + ((lane >> 4) * 16);
#pragma unroll
  for (int i = 0; i < 16; ++i) {
    int kk = koff + i;
    a[i] = (kk < kmax) ? (_Float16)(At[(size_t)kk * ld + row] * scale)
                       : (_Float16)0.0f;
  }
  return a;
}

// B: 32x16 f16. lane L: col = L&15; half i -> K = (L>>4)*16 + i
__device__ __forceinline__ v16h bfrag(const float* Bm, int ldb, int col0, int k0,
                                      int lane) {
  v16h b;
  const float* bp = Bm + (size_t)(k0 + ((lane >> 4) * 16)) * ldb + col0 + (lane & 15);
#pragma unroll
  for (int i = 0; i < 16; ++i) b[i] = (_Float16)bp[(size_t)i * ldb];
  return b;
}

__device__ __forceinline__ v16h bfrag_m(const float* Bm, int ldb, int col0,
                                        int k0, int kmax, int lane) {
  v16h b;
  int col = col0 + (lane & 15);
  int kb = k0 + ((lane >> 4) * 16);
#pragma unroll
  for (int i = 0; i < 16; ++i) {
    int kk = kb + i;
    b[i] = (kk < kmax) ? (_Float16)Bm[(size_t)kk * ldb + col] : (_Float16)0.0f;
  }
  return b;
}

// transposed B source: B[kk][n] = Bt[(col0+n)*ldn + kk] (contiguous along K)
__device__ __forceinline__ v16h bfragT(const float* Bt, int ldn, int col0, int k0,
                                       int lane) {
  v16h b;
  const float* bp = Bt + (size_t)(col0 + (lane & 15)) * ldn + k0 + ((lane >> 4) * 16);
#pragma unroll
  for (int i = 0; i < 16; ++i) b[i] = (_Float16)bp[i];
  return b;
}

// C/D: 16x16 f32. lane L: col = L&15; elem r -> row = r + (L>>4)*8
__device__ __forceinline__ v8f cfrag_load(const float* Cm, int ldc, int row0,
                                          int col0, int lane) {
  v8f c;
  const float* cp = Cm + (size_t)(row0 + ((lane >> 4) * 8)) * ldc + col0 + (lane & 15);
#pragma unroll
  for (int r = 0; r < 8; ++r) c[r] = cp[(size_t)r * ldc];
  return c;
}

__device__ __forceinline__ void cfrag_store(float* Dm, int ldc, int row0, int col0,
                                            int lane, v8f acc) {
  float* dp = Dm + (size_t)(row0 + ((lane >> 4) * 8)) * ldc + col0 + (lane & 15);
#pragma unroll
  for (int r = 0; r < 8; ++r) dp[(size_t)r * ldc] = acc[r];
}

// ---- kernels ---------------------------------------------------------------

// q = per-group 1x1 conv: Wq[g](64x64) @ X[b,g](64x1024). 8 waves/block.
// Weight tile async-staged to LDS via GLOBAL_LOAD_ASYNC_TO_LDS_B128 (ASYNCcnt).
__global__ void k_qproj(const float* __restrict__ x, const float* __restrict__ q_w,
                        float* __restrict__ q) {
  __shared__ float Alds[kCG * kCG];
  int lane = threadIdx.x & 31;
  int t = blockIdx.x * (blockDim.x >> 5) + (threadIdx.x >> 5);  // 4096 tiles
  int bg = t >> 8;
  int r = t & 255;
  int mt = r >> 6, nt = r & 63;
  int g = bg & (kG - 1);
  const float* A = q_w + (size_t)g * kCG * kCG;
  // stage 16KB of weights: 1024 x 16B async copies, 4 per thread
  for (int idx = threadIdx.x; idx < (kCG * kCG / 4); idx += blockDim.x) {
    unsigned lo = (unsigned)(size_t)(Alds + idx * 4);
    unsigned long long ga = (unsigned long long)(size_t)(A + idx * 4);
    asm volatile("global_load_async_to_lds_b128 %0, %1, off"
                 :: "v"(lo), "v"(ga) : "memory");
  }
  asm volatile("s_wait_asynccnt 0x0" ::: "memory");
  __syncthreads();

  const float* Bm = x + (size_t)bg * kCG * kHW;
  float* D = q + (size_t)bg * kCG * kHW;
  v8f acc = {};
#pragma unroll
  for (int k0 = 0; k0 < kCG; k0 += 32) {
    acc = wmma16x16x32(afrag(Alds, kCG, mt * 16, k0, lane),
                       bfrag(Bm, kHW, nt * 16, k0, lane), acc);
  }
  cfrag_store(D, kHW, mt * 16, nt * 16, lane, acc);
}

// offset net: depthwise 6x6 stride4 conv + GELU + 1x1(64->2) + tanh*4, then
// build normalized deformed grid directly. 1 block per (b,g), thread = pixel.
__global__ void k_offset(const float* __restrict__ q, const float* __restrict__ dw_w,
                         const float* __restrict__ dw_b, const float* __restrict__ pw_w,
                         float* __restrict__ vgrid) {
  int bg = blockIdx.x;
  int p = threadIdx.x;  // 0..63
  int oy = p >> 3, ox = p & 7;
  const float* qb = q + (size_t)bg * kCG * kHW;
  float px = 0.f, py = 0.f;
  for (int c = 0; c < kCG; ++c) {
    float acc = dw_b[c];
    const float* wc = dw_w + c * kKOFF * kKOFF;
    const float* qc = qb + (size_t)c * kHW;
#pragma unroll
    for (int ky = 0; ky < kKOFF; ++ky) {
      int iy = oy * kDF - 1 + ky;
      if (iy < 0 || iy >= kH) continue;
#pragma unroll
      for (int kx = 0; kx < kKOFF; ++kx) {
        int ix = ox * kDF - 1 + kx;
        if (ix < 0 || ix >= kW) continue;
        acc += wc[ky * kKOFF + kx] * qc[iy * kW + ix];
      }
    }
    float gelu = 0.5f * acc * (1.0f + erff(acc * 0.70710678118654752f));
    px += pw_w[c] * gelu;
    py += pw_w[kCG + c] * gelu;
  }
  float gx = (float)ox + tanhf(px) * kOSCALE;
  float gy = (float)oy + tanhf(py) * kOSCALE;
  vgrid[(bg * kNJ + p) * 2 + 0] = 2.0f * gx / (float)(kH2 - 1) - 1.0f;
  vgrid[(bg * kNJ + p) * 2 + 1] = 2.0f * gy / (float)(kW2 - 1) - 1.0f;
}

// bilinear grid_sample (align_corners=False, zero pad): kv[bg,c,j]
__global__ void k_sample(const float* __restrict__ x, const float* __restrict__ vgrid,
                         float* __restrict__ kv) {
  int idx = blockIdx.x * blockDim.x + threadIdx.x;  // 65536
  int bg = idx >> 12;
  int c = (idx >> 6) & 63;
  int j = idx & 63;
  float gx = vgrid[(bg * kNJ + j) * 2 + 0];
  float gy = vgrid[(bg * kNJ + j) * 2 + 1];
  float fx = (gx + 1.0f) * (kW * 0.5f) - 0.5f;
  float fy = (gy + 1.0f) * (kH * 0.5f) - 0.5f;
  float x0 = floorf(fx), y0 = floorf(fy);
  float wx = fx - x0, wy = fy - y0;
  int x0i = (int)x0, y0i = (int)y0;
  const float* img = x + (size_t)(bg * kCG + c) * kHW;
  auto tap = [&](int xi, int yi) -> float {
    if (xi < 0 || xi >= kW || yi < 0 || yi >= kH) return 0.f;
    return img[yi * kW + xi];
  };
  float g00 = tap(x0i, y0i), g01 = tap(x0i + 1, y0i);
  float g10 = tap(x0i, y0i + 1), g11 = tap(x0i + 1, y0i + 1);
  kv[idx] = g00 * (1 - wx) * (1 - wy) + g01 * wx * (1 - wy) +
            g10 * (1 - wx) * wy + g11 * wx * wy;
}

// k/v grouped projections: Wk/Wv[g](64x64) @ kv[bg](64x64)
__global__ void k_kvproj(const float* __restrict__ kv, const float* __restrict__ k_w,
                         const float* __restrict__ v_w, float* __restrict__ kmat,
                         float* __restrict__ vmat) {
  int lane = threadIdx.x & 31;
  int t = blockIdx.x * (blockDim.x >> 5) + (threadIdx.x >> 5);  // 512 tiles
  int bg = t >> 5;
  int r = t & 31;
  int isV = r >> 4;
  int rr = r & 15;
  int mt = rr >> 2, nt = rr & 3;
  int g = bg & (kG - 1);
  const float* A = (isV ? v_w : k_w) + (size_t)g * kCG * kCG;
  const float* Bm = kv + (size_t)bg * kCG * kNJ;
  float* D = (isV ? vmat : kmat) + (size_t)bg * kCG * kNJ;
  v8f acc = {};
#pragma unroll
  for (int k0 = 0; k0 < kCG; k0 += 32) {
    acc = wmma16x16x32(afrag(A, kCG, mt * 16, k0, lane),
                       bfrag(Bm, kNJ, nt * 16, k0, lane), acc);
  }
  cfrag_store(D, kNJ, mt * 16, nt * 16, lane, acc);
}

// CPB MLP: per (bg, i) process all 64 j. Layer1 (2->64) VALU into LDS (f16 in
// A-frag order), Layer2 (64->64) via WMMA with W2 staged to LDS by the
// Tensor Data Mover (TENSOR_LOAD_TO_LDS + s_wait_tensorcnt), Layer3 (64->4)
// per-lane dot. Writes bias directly into the sim buffer (b,16,1024,64).
__global__ void k_cpb(const float* __restrict__ vgrid, const float* __restrict__ w1,
                      const float* __restrict__ b1, const float* __restrict__ w2,
                      const float* __restrict__ b2, const float* __restrict__ w3,
                      const float* __restrict__ b3, float* __restrict__ simb) {
  __shared__ float posb[kNJ][2];
  __shared__ _Float16 h1[kNJ * kCPB];
  __shared__ float h2buf[kNJ * kCPB];
  __shared__ float w2lds[kCPB * kCPB];
  int bg = blockIdx.x >> 10;
  int i = blockIdx.x & 1023;
  int tid = threadIdx.x;

  // TDM: DMA the 64x64 f32 W2 tensor into LDS (one descriptor, wave 0 only).
  if (tid < 32) {
    unsigned ldsa = (unsigned)(size_t)w2lds;             // LDS byte offset
    unsigned long long ga = (unsigned long long)(size_t)w2;
    v4u g0;
    g0[0] = 1u;                                          // count=1, user mode
    g0[1] = ldsa;                                        // lds_addr
    g0[2] = (unsigned)ga;                                // global_addr[31:0]
    g0[3] = ((unsigned)(ga >> 32) & 0x01FFFFFFu) | 0x80000000u;  // [56:32]|type=2
    v8u g1;
    g1[0] = 2u << 16;   // data_size = 4B; wg_mask=0; no pad/iterate/barrier
    g1[1] = 64u << 16;  // tensor_dim0 = 64    (bits 79:48)
    g1[2] = 64u << 16;  // tensor_dim1 = 64    (bits 111:80)
    g1[3] = 64u << 16;  // tile_dim0   = 64    (bits 127:112)
    g1[4] = 64u;        // tile_dim1   = 64; tile_dim2 = 0
    g1[5] = 64u;        // tensor_dim0_stride = 64
    g1[6] = 0u;
    g1[7] = 0u;
    asm volatile("tensor_load_to_lds %0, %1" :: "s"(g0), "s"(g1) : "memory");
    asm volatile("s_wait_tensorcnt 0x0" ::: "memory");
  }

  if (tid < kNJ) {
    int j = tid;
    float vx = vgrid[(bg * kNJ + j) * 2 + 0];
    float vy = vgrid[(bg * kNJ + j) * 2 + 1];
    float qxn = 2.0f * (float)(i & 31) / 31.0f - 1.0f;
    float qyn = 2.0f * (float)(i >> 5) / 31.0f - 1.0f;
    float pxv = qxn - vx, pyv = qyn - vy;
    posb[j][0] = copysignf(log1pf(fabsf(pxv)), pxv);
    posb[j][1] = copysignf(log1pf(fabsf(pyv)), pyv);
  }
  __syncthreads();
  for (int idx = tid; idx < kNJ * kCPB; idx += blockDim.x) {
    int j = idx >> 6, c = idx & 63;
    float v = w1[c * 2 + 0] * posb[j][0] + w1[c * 2 + 1] * posb[j][1] + b1[c];
    h1[j * kCPB + c] = (_Float16)(v > 0.f ? v : 0.f);
  }
  __syncthreads();

  int wave = tid >> 5, lane = tid & 31;
#pragma unroll
  for (int ntile = 0; ntile < 4; ++ntile) {
    v8f acc = {};
#pragma unroll
    for (int k0 = 0; k0 < kCPB; k0 += 32) {
      acc = wmma16x16x32(afrag_h(h1, kCPB, wave * 16, k0, lane),
                         bfragT(w2lds, kCPB, ntile * 16, k0, lane), acc);
    }
    int col = ntile * 16 + (lane & 15);
    int rb = (lane >> 4) * 8;
    float bias2 = b2[col];
#pragma unroll
    for (int rr = 0; rr < 8; ++rr) {
      float v = acc[rr] + bias2;
      h2buf[(wave * 16 + rb + rr) * kCPB + col] = v > 0.f ? v : 0.f;
    }
  }
  __syncthreads();

  int b_ = bg >> 2, g = bg & 3;
  for (int o = lane; o < 64; o += 32) {
    int tt = o >> 2, hp = o & 3;
    const float* h2r = h2buf + (wave * 16 + tt) * kCPB;
    float s = b3[hp];
    for (int c = 0; c < kCPB; ++c) s += h2r[c] * w3[hp * kCPB + c];
    int head = g * kHPG + hp;
    int j = wave * 16 + tt;
    simb[((size_t)(b_ * kHEADS + head) * kHW + i) * kNJ + j] = s;
  }
}

// sim += (q*scale)^T @ k : K=16 zero-padded to 32; C-operand accumulates the
// CPB bias already in simb.
__global__ void k_sim(const float* __restrict__ q, const float* __restrict__ kmat,
                      float* __restrict__ simb) {
  int lane = threadIdx.x & 31;
  int t = blockIdx.x * (blockDim.x >> 5) + (threadIdx.x >> 5);  // 16384 tiles
  int bh = t >> 8;
  int r = t & 255;
  int it = r >> 2, jt = r & 3;
  const float* qb = q + (size_t)bh * kDH * kHW;
  const float* kb = kmat + (size_t)bh * kDH * kNJ;
  float* sp = simb + (size_t)bh * kHW * kNJ;
  v8f acc = cfrag_load(sp, kNJ, it * 16, jt * 16, lane);
  v16h a = afragT_ms(qb, kHW, it * 16, 0, kDH, 0.25f, lane);  // DH^-0.5
  v16h b = bfrag_m(kb, kNJ, jt * 16, 0, kDH, lane);
  acc = wmma16x16x32(a, b, acc);
  cfrag_store(sp, kNJ, it * 16, jt * 16, lane, acc);
}

// softmax over j=64, one wave per row, wave32 shuffles
__global__ void k_softmax(float* __restrict__ simb) {
  int lane = threadIdx.x & 31;
  int row = blockIdx.x * (blockDim.x >> 5) + (threadIdx.x >> 5);  // 65536 rows
  float* p = simb + (size_t)row * kNJ;
  float v0 = p[lane], v1 = p[lane + 32];
  float m = fmaxf(v0, v1);
#pragma unroll
  for (int off = 16; off; off >>= 1) m = fmaxf(m, __shfl_xor(m, off, 32));
  float e0 = __expf(v0 - m), e1 = __expf(v1 - m);
  float s = e0 + e1;
#pragma unroll
  for (int off = 16; off; off >>= 1) s += __shfl_xor(s, off, 32);
  float inv = 1.0f / s;
  p[lane] = e0 * inv;
  p[lane + 32] = e1 * inv;
}

// outh = attn(1024x64) @ v^T(64x16) per (b,head)
__global__ void k_attnv(const float* __restrict__ simb, const float* __restrict__ vmat,
                        float* __restrict__ outh) {
  int lane = threadIdx.x & 31;
  int t = blockIdx.x * (blockDim.x >> 5) + (threadIdx.x >> 5);  // 4096 tiles
  int bh = t >> 6;
  int it = t & 63;
  const float* A = simb + ((size_t)bh * kHW + it * 16) * kNJ;
  const float* Vt = vmat + (size_t)bh * kDH * kNJ;
  float* D = outh + (size_t)bh * kDH * kHW;
  v8f acc = {};
#pragma unroll
  for (int k0 = 0; k0 < kNJ; k0 += 32) {
    acc = wmma16x16x32(afrag(A, kNJ, 0, k0, lane), bfragT(Vt, kNJ, 0, k0, lane),
                       acc);
  }
  // D[m][n]: channel = n (col), position = it*16 + m (row)
  float* dp = D + (size_t)(lane & 15) * kHW + it * 16 + ((lane >> 4) * 8);
#pragma unroll
  for (int rr = 0; rr < 8; ++rr) dp[rr] = acc[rr];
}

// out = out_w(256x256) @ outh(256x1024) + out_b, per batch
__global__ void k_outproj(const float* __restrict__ outh, const float* __restrict__ out_w,
                          const float* __restrict__ out_b, float* __restrict__ proj) {
  int lane = threadIdx.x & 31;
  int t = blockIdx.x * (blockDim.x >> 5) + (threadIdx.x >> 5);  // 4096 tiles
  int b_ = t >> 10;
  int r = t & 1023;
  int mt = r >> 6, nt = r & 63;
  const float* Bm = outh + (size_t)b_ * kC * kHW;
  v8f acc = {};
#pragma unroll
  for (int k0 = 0; k0 < kC; k0 += 32) {
    if (k0 + 32 < kC)  // prefetch next B chunk (global_prefetch_b8)
      __builtin_prefetch(Bm + (size_t)(k0 + 32) * kHW + nt * 16 + lane, 0, 1);
    acc = wmma16x16x32(afrag(out_w, kC, mt * 16, k0, lane),
                       bfrag(Bm, kHW, nt * 16, k0, lane), acc);
  }
  int rb = mt * 16 + ((lane >> 4) * 8);
  int col = nt * 16 + (lane & 15);
  float* dp = proj + ((size_t)b_ * kC + rb) * kHW + col;
#pragma unroll
  for (int rr = 0; rr < 8; ++rr) dp[(size_t)rr * kHW] = acc[rr] + out_b[rb + rr];
}

// residual + LayerNorm over C=256: one wave per (b, spatial)
__global__ void k_ln(const float* __restrict__ proj, const float* __restrict__ x,
                     const float* __restrict__ ln_g, const float* __restrict__ ln_b,
                     float* __restrict__ out) {
  int lane = threadIdx.x & 31;
  int row = blockIdx.x * (blockDim.x >> 5) + (threadIdx.x >> 5);  // 4096 rows
  int b_ = row >> 10;
  int i = row & 1023;
  const float* pb_ = proj + (size_t)b_ * kC * kHW + i;
  const float* xb = x + (size_t)b_ * kC * kHW + i;
  float y[8];
  float s = 0.f, s2 = 0.f;
#pragma unroll
  for (int kk = 0; kk < 8; ++kk) {
    int c = lane + kk * 32;
    float v = pb_[(size_t)c * kHW] + xb[(size_t)c * kHW];
    y[kk] = v;
    s += v;
    s2 += v * v;
  }
#pragma unroll
  for (int off = 16; off; off >>= 1) {
    s += __shfl_xor(s, off, 32);
    s2 += __shfl_xor(s2, off, 32);
  }
  float mu = s * (1.0f / kC);
  float var = s2 * (1.0f / kC) - mu * mu;
  float rs = rsqrtf(var + 1e-5f);
  float* ob = out + (size_t)b_ * kC * kHW + i;
#pragma unroll
  for (int kk = 0; kk < 8; ++kk) {
    int c = lane + kk * 32;
    ob[(size_t)c * kHW] = (y[kk] - mu) * rs * ln_g[c] + ln_b[c];
  }
}

}  // namespace

extern "C" void kernel_launch(void* const* d_in, const int* in_sizes, int n_in,
                              void* d_out, int out_size, void* d_ws, size_t ws_size,
                              hipStream_t stream) {
  (void)in_sizes; (void)n_in; (void)out_size; (void)ws_size;
  const float* x      = (const float*)d_in[0];
  const float* q_w    = (const float*)d_in[1];
  const float* k_w    = (const float*)d_in[2];
  const float* v_w    = (const float*)d_in[3];
  const float* out_w  = (const float*)d_in[4];
  const float* out_b  = (const float*)d_in[5];
  const float* dw_w   = (const float*)d_in[6];
  const float* dw_b   = (const float*)d_in[7];
  const float* pw_w   = (const float*)d_in[8];
  const float* cpb_w1 = (const float*)d_in[9];
  const float* cpb_b1 = (const float*)d_in[10];
  const float* cpb_w2 = (const float*)d_in[11];
  const float* cpb_b2 = (const float*)d_in[12];
  const float* cpb_w3 = (const float*)d_in[13];
  const float* cpb_b3 = (const float*)d_in[14];
  const float* ln_g   = (const float*)d_in[15];
  const float* ln_b   = (const float*)d_in[16];
  float* out = (float*)d_out;

  float* ws = (float*)d_ws;
  float* q     = ws;  ws += (size_t)kB * kC * kHW;           // 1M
  float* kv    = ws;  ws += (size_t)kB * kG * kCG * kNJ;     // 64K
  float* kmat  = ws;  ws += (size_t)kB * kC * kNJ;           // 64K
  float* vmat  = ws;  ws += (size_t)kB * kC * kNJ;           // 64K
  float* simb  = ws;  ws += (size_t)kB * kHEADS * kHW * kNJ; // 4M
  float* outh  = ws;  ws += (size_t)kB * kC * kHW;           // 1M
  float* proj  = ws;  ws += (size_t)kB * kC * kHW;           // 1M
  float* vgrid = ws;  ws += (size_t)kB * kG * kNJ * 2;       // 2K

  k_qproj<<<512, 256, 0, stream>>>(x, q_w, q);
  k_offset<<<kB * kG, 64, 0, stream>>>(q, dw_w, dw_b, pw_w, vgrid);
  k_sample<<<256, 256, 0, stream>>>(x, vgrid, kv);
  k_kvproj<<<64, 256, 0, stream>>>(kv, k_w, v_w, kmat, vmat);
  k_cpb<<<kB * kG * kHW, 128, 0, stream>>>(vgrid, cpb_w1, cpb_b1, cpb_w2, cpb_b2,
                                           cpb_w3, cpb_b3, simb);
  k_sim<<<2048, 256, 0, stream>>>(q, kmat, simb);
  k_softmax<<<8192, 256, 0, stream>>>(simb);
  k_attnv<<<512, 256, 0, stream>>>(simb, vmat, outh);
  k_outproj<<<512, 256, 0, stream>>>(outh, out_w, out_b, proj);
  k_ln<<<512, 256, 0, stream>>>(proj, x, ln_g, ln_b, out);
}
